// EigenBasisEncoder_56212531970552
// MI455X (gfx1250) — matrix operand order repaired and checked by reading, not compiled, not run on hardware
//
#include <hip/hip_runtime.h>

typedef __attribute__((ext_vector_type(16))) _Float16 v16h;
typedef __attribute__((ext_vector_type(8)))  float    v8f;

#define B_   4
#define M_   16
#define NN_  16384   // N*N = 128*128
#define D_   32
#define HF_  256
#define HG_  64

// Fast silu: z * rcp(1 + exp(-z)).  v_exp_f32 + v_add + v_rcp_f32 + v_mul.
__device__ __forceinline__ float silu_f(float z) {
  return z * __builtin_amdgcn_rcpf(1.0f + __expf(-z));
}

// ---------------------------------------------------------------------------
// Kernel 1: f[b,m,d] = exp( x*w1[d]+b1[d] + sum_h silu(x*w2[h]+b2[h])*w3[h][d] ) * mask_m
// 64 (b,m) scalars -> one wave each, lane = output channel d.
// ---------------------------------------------------------------------------
__global__ void __launch_bounds__(256)
eigval_f_kernel(const float* __restrict__ eigval, const float* __restrict__ mask_m,
                const float* __restrict__ w1, const float* __restrict__ b1,
                const float* __restrict__ w2, const float* __restrict__ b2,
                const float* __restrict__ w3, float* __restrict__ fout) {
  int gid = blockIdx.x * 256 + threadIdx.x;   // 2048 threads = 64 waves
  int bm  = gid >> 5;                         // 0..63 = b*16+m
  int d   = gid & 31;                         // 0..31 = D_
  float x = eigval[bm];
  float acc = fmaf(x, w1[d], b1[d]);
  #pragma unroll 8
  for (int h = 0; h < HF_; ++h) {
    float z = fmaf(x, w2[h], b2[h]);
    acc = fmaf(silu_f(z), w3[h * D_ + d], acc);
  }
  fout[bm * D_ + d] = __expf(acc) * mask_m[bm];
}

// ---------------------------------------------------------------------------
// Kernel 2: fused  g = silu(affine + HH@W3),  out = sum_m f*g,  y = edge + out@linW^T + linb
// One wave per 16-position tile. All matmuls on v_wmma_f32_16x16x32_f16.
// The two 16-wide d-tiles use interleaved accumulator chains so dependent
// WMMAs are separated by an independent WMMA; fragment build + consume stay
// in-iteration (the scheduler interleaves them with the WMMAs best this way).
// ---------------------------------------------------------------------------
__global__ void __launch_bounds__(256)
eigvec_main_kernel(const float* __restrict__ eigvec,
                   const float* __restrict__ edge_attr,
                   const float* __restrict__ g1w, const float* __restrict__ g1b,
                   const float* __restrict__ g2w, const float* __restrict__ g2b,
                   const float* __restrict__ g3w,
                   const float* __restrict__ linw, const float* __restrict__ linb,
                   const float* __restrict__ ftab,
                   float* __restrict__ out) {
  __shared__ float    fl[M_ * D_];             // f[b] table: 16x32 f32 (2 KB)
  __shared__ _Float16 xpose[8][16 * D_];       // per-wave C->A transpose staging (8 KB)

  const int tid  = threadIdx.x;
  const int wave = tid >> 5;
  const int lane = tid & 31;
  const int b    = blockIdx.x >> 7;                      // 128 blocks per batch
  const int tile = ((blockIdx.x & 127) << 3) + wave;     // 0..1023 within batch
  const int pos_base = tile << 4;                        // 16 positions per tile
  const int row = lane & 15;                             // A-fragment row (M index)
  const int hi  = lane >> 4;                             // lane half
  const int n16 = lane & 15;                             // B/C column within d-tile

  const size_t obase = ((size_t)b * NN_ + pos_base) * D_;

  // prefetch this tile's edge_attr slice (2 KB, used only in the epilogue)
  __builtin_prefetch(edge_attr + obase + (size_t)lane * 16, 0, 0);

  // ---- stage f[b,:,:] into LDS (512 floats, 256 threads x 2) ----
  fl[tid]       = ftab[b * (M_ * D_) + tid];
  fl[tid + 256] = ftab[b * (M_ * D_) + 256 + tid];
  __syncthreads();

  // ---- 16-bit A-fragment K map (contiguous, matches B: k = hi*16 + e) ----
  int klA[16];
  #pragma unroll
  for (int v = 0; v < 8; ++v) {
    int base = hi * 16 + v * 2;
    klA[2 * v]     = base;
    klA[2 * v + 1] = base + 1;
  }
  // 16-bit B-fragment K map (32x16: lanes 0-15 hold K 0..15, lanes 16-31 hold
  // K 16..31): k_local_B(e) = hi*16 + e

  // ---- per-lane hidden weights in A-K order, both K tiles ----
  float w2r[32], b2r[32];
  #pragma unroll
  for (int t = 0; t < 2; ++t)
    #pragma unroll
    for (int e = 0; e < 16; ++e) {
      int k = t * 32 + klA[e];
      w2r[t * 16 + e] = g2w[k];
      b2r[t * 16 + e] = g2b[k];
    }

  // ---- B fragments: W3 (64x32) as 2 K-tiles x 2 d-tiles ----
  v16h bg[2][2];
  #pragma unroll
  for (int t = 0; t < 2; ++t)
    #pragma unroll
    for (int nt = 0; nt < 2; ++nt)
      #pragma unroll
      for (int e = 0; e < 16; ++e)
        bg[t][nt][e] = (_Float16)g3w[(t * 32 + hi * 16 + e) * D_ + nt * 16 + n16];

  // ---- extension B fragment: rows [w1 ; b1] folds x*w1+b1 into the GEMM ----
  v16h bext[2];
  #pragma unroll
  for (int nt = 0; nt < 2; ++nt)
    #pragma unroll
    for (int e = 0; e < 16; ++e) {
      float wv = (hi == 0 && e == 0) ? g1w[nt * 16 + n16]
               : (hi == 0 && e == 1) ? g1b[nt * 16 + n16] : 0.0f;
      bext[nt][e] = (_Float16)wv;
    }

  // ---- B fragments for final linear: B[k=d][n=o] = linw[o][d], K = 32 ----
  v16h blin[2];
  #pragma unroll
  for (int nt = 0; nt < 2; ++nt)
    #pragma unroll
    for (int e = 0; e < 16; ++e)
      blin[nt][e] = (_Float16)linw[(nt * 16 + n16) * D_ + hi * 16 + e];

  // ---- preload eigvec scalars for all 16 m (row owned by this lane) ----
  const float* evb = eigvec + ((size_t)b * M_) * NN_ + pos_base + row;
  float xv[16];
  #pragma unroll
  for (int m = 0; m < M_; ++m) xv[m] = evb[(size_t)m * NN_];

  v8f acc[2];
  #pragma unroll
  for (int nt = 0; nt < 2; ++nt)
    #pragma unroll
    for (int r = 0; r < 8; ++r) acc[nt][r] = 0.0f;

  // ================= m-reduction: 16 x (silu-hidden -> 6 interleaved WMMAs) =
  #pragma unroll 4
  for (int m = 0; m < M_; ++m) {
    float x = xv[m];
    v16h a0, a1, ax;
    #pragma unroll
    for (int e = 0; e < 16; ++e) {
      float z0 = fmaf(x, w2r[e],      b2r[e]);
      float z1 = fmaf(x, w2r[16 + e], b2r[16 + e]);
      a0[e] = (_Float16)silu_f(z0);
      a1[e] = (_Float16)silu_f(z1);
      float av = (hi == 0 && e == 0) ? x : (hi == 0 && e == 1) ? 1.0f : 0.0f;
      ax[e] = (_Float16)av;
    }
    v8f c0, c1;
    #pragma unroll
    for (int r = 0; r < 8; ++r) { c0[r] = 0.0f; c1[r] = 0.0f; }
    // interleaved chains: each dependent WMMA has an independent one between
    c0 = __builtin_amdgcn_wmma_f32_16x16x32_f16(false, a0, false, bg[0][0], (short)0, c0, false, false);
    c1 = __builtin_amdgcn_wmma_f32_16x16x32_f16(false, a0, false, bg[0][1], (short)0, c1, false, false);
    c0 = __builtin_amdgcn_wmma_f32_16x16x32_f16(false, a1, false, bg[1][0], (short)0, c0, false, false);
    c1 = __builtin_amdgcn_wmma_f32_16x16x32_f16(false, a1, false, bg[1][1], (short)0, c1, false, false);
    c0 = __builtin_amdgcn_wmma_f32_16x16x32_f16(false, ax, false, bext[0],  (short)0, c0, false, false);
    c1 = __builtin_amdgcn_wmma_f32_16x16x32_f16(false, ax, false, bext[1],  (short)0, c1, false, false);

    float f0 = fl[m * D_ + n16];            // ds_load per (m, d-tile)
    float f1 = fl[m * D_ + 16 + n16];
    #pragma unroll
    for (int r = 0; r < 8; ++r) {
      acc[0][r] = fmaf(f0, silu_f(c0[r]), acc[0][r]);
      acc[1][r] = fmaf(f1, silu_f(c1[r]), acc[1][r]);
    }
  }

  // ====== C-layout -> A-layout transpose through per-wave LDS (fp16) =======
  _Float16* xb = &xpose[wave][0];
  #pragma unroll
  for (int nt = 0; nt < 2; ++nt)
    #pragma unroll
    for (int r = 0; r < 8; ++r)
      xb[(r + hi * 8) * D_ + nt * 16 + n16] = (_Float16)acc[nt][r];

  asm volatile("s_wait_dscnt 0" ::: "memory");   // LDS stores visible before reload

  v16h alin;
  #pragma unroll
  for (int v = 0; v < 8; ++v) {
    int k = klA[2 * v];
    alin[2 * v]     = xb[row * D_ + k];
    alin[2 * v + 1] = xb[row * D_ + k + 1];
  }

  // ====== final linear: y = edge_attr + lin_b + acc @ linW^T (1 WMMA/d-tile)
  v8f ci0, ci1;
  #pragma unroll
  for (int r = 0; r < 8; ++r) {
    ci0[r] = edge_attr[obase + (size_t)(r + hi * 8) * D_ + n16]      + linb[n16];
    ci1[r] = edge_attr[obase + (size_t)(r + hi * 8) * D_ + 16 + n16] + linb[16 + n16];
  }
  v8f y0 = __builtin_amdgcn_wmma_f32_16x16x32_f16(false, alin, false, blin[0], (short)0, ci0, false, false);
  v8f y1 = __builtin_amdgcn_wmma_f32_16x16x32_f16(false, alin, false, blin[1], (short)0, ci1, false, false);
  #pragma unroll
  for (int r = 0; r < 8; ++r) {
    out[obase + (size_t)(r + hi * 8) * D_ + n16]      = y0[r];
    out[obase + (size_t)(r + hi * 8) * D_ + 16 + n16] = y1[r];
  }
}

// ---------------------------------------------------------------------------
extern "C" void kernel_launch(void* const* d_in, const int* in_sizes, int n_in,
                              void* d_out, int out_size, void* d_ws, size_t ws_size,
                              hipStream_t stream) {
  const float* eigval    = (const float*)d_in[0];
  const float* eigvec    = (const float*)d_in[1];
  const float* mask_m    = (const float*)d_in[2];
  // d_in[3] = mask_all (all-true bool) -- no-op in this graph, skipped
  const float* edge_attr = (const float*)d_in[4];
  const float* f1w = (const float*)d_in[5];
  const float* f1b = (const float*)d_in[6];
  const float* f2w = (const float*)d_in[7];
  const float* f2b = (const float*)d_in[8];
  const float* f3w = (const float*)d_in[9];
  const float* g1w = (const float*)d_in[10];
  const float* g1b = (const float*)d_in[11];
  const float* g2w = (const float*)d_in[12];
  const float* g2b = (const float*)d_in[13];
  const float* g3w = (const float*)d_in[14];
  const float* linw = (const float*)d_in[15];
  const float* linb = (const float*)d_in[16];

  float* ftab = (float*)d_ws;   // 64 x 32 f32 = 8 KB scratch

  // f(eigval): 64 waves
  eigval_f_kernel<<<8, 256, 0, stream>>>(eigval, mask_m, f1w, f1b, f2w, f2b, f3w, ftab);

  // fused g / m-sum / linear: 4096 tiles, 8 waves per block
  eigvec_main_kernel<<<512, 256, 0, stream>>>(eigvec, edge_attr,
                                              g1w, g1b, g2w, g2b, g3w,
                                              linw, linb, ftab, (float*)d_out);
}
